// MultiHeadAttention_17059610100278
// MI455X (gfx1250) — matrix operand order repaired and checked by reading, not compiled, not run on hardware
//
#include <hip/hip_runtime.h>
#include <hip/hip_bf16.h>

// MHA forward for MI455X (gfx1250, wave32). bf16 WMMA with fp32 accumulation.
// D_MODEL=2048, N_HEADS=16, D_HEAD=128, B=2, S=2048.

#define DM   2048
#define NH   16
#define DH   128
#define SEQ  2048
#define BATCH 2

typedef __attribute__((ext_vector_type(16))) __bf16       bf16x16;
typedef __attribute__((ext_vector_type(8)))  float        f32x8;
typedef __attribute__((ext_vector_type(4)))  float        f32x4;
typedef __attribute__((ext_vector_type(4)))  unsigned int u32x4;

union FragAB { bf16x16 v; u32x4 q[2]; };

__device__ __forceinline__ unsigned short f2bf(float f) {
  union { float f; unsigned int u; } c; c.f = f;
  unsigned int u = c.u + 0x7FFFu + ((c.u >> 16) & 1u);   // round-to-nearest-even
  return (unsigned short)(u >> 16);
}
__device__ __forceinline__ float bf2f(unsigned short h) {
  union { unsigned int u; float f; } c; c.u = ((unsigned int)h) << 16;
  return c.f;
}

// gfx1250 async memory->LDS copy (16B per lane), tracked with ASYNCcnt.
// VDST supplies the per-lane LDS byte address; generic pointers to LDS carry
// the LDS offset in their low 32 bits (ISA 10.2).
__device__ __forceinline__ void async_load_b128(void* lds_ptr, const void* gptr) {
  unsigned lds_off = (unsigned)(uintptr_t)lds_ptr;
  unsigned long long ga = (unsigned long long)(uintptr_t)gptr;
  asm volatile("global_load_async_to_lds_b128 %0, %1, off"
               :: "v"(lds_off), "v"(ga) : "memory");
}
__device__ __forceinline__ void wait_async0() {
  asm volatile("s_wait_asynccnt 0x0" ::: "memory");
}

// ---------------------------------------------------------------------------
// Tiled GEMM, C = A(fp32)[M,K] * B(fp32)[K,N] + bias, via bf16 WMMA, fp32 acc.
// Tile: 128x128x32, 256 threads = 8 waves in a 4(m) x 2(n) grid; each wave
// owns a 32x64 patch = 2x4 v_wmma_f32_16x16x32_bf16 accumulators.
// Register double-buffered: next tile's global loads overlap current WMMAs.
// MODE 0: epilogue scatters to bf16 Q/K/V in [B,H,S,DH] layout (QKV proj).
// MODE 1: epilogue writes fp32 row-major out (output projection).
// ---------------------------------------------------------------------------
template<int MODE>
__global__ __launch_bounds__(256) void gemm_bf16_wmma(
    const float* __restrict__ A, const float* __restrict__ Bm,
    const float* __restrict__ bias,
    unsigned short* __restrict__ qB, unsigned short* __restrict__ kB,
    unsigned short* __restrict__ vB, float* __restrict__ out,
    int N, int K)
{
  // +8 ushort pad -> 80B row stride: fragment lanes land on distinct banks.
  __shared__ __align__(16) unsigned short As[128][40]; // A tile, [m][k]
  __shared__ __align__(16) unsigned short Bs[128][40]; // B tile transposed, [n][k]

  const int t    = threadIdx.x;
  const int lane = t & 31;
  const int wid  = t >> 5;
  const int half = lane >> 4;
  const int l16  = lane & 15;
  const int wm   = wid & 3;        // wave row  (32 rows each)
  const int wn   = wid >> 2;       // wave col  (64 cols each)
  const int m0   = blockIdx.y * 128;
  const int n0   = blockIdx.x * 128;

  const f32x8 zero8 = {0.f,0.f,0.f,0.f,0.f,0.f,0.f,0.f};
  f32x8 acc[2][4];
#pragma unroll
  for (int i = 0; i < 2; ++i)
#pragma unroll
    for (int j = 0; j < 4; ++j) acc[i][j] = zero8;

  const int a_r = t >> 3,  a_c = (t & 7)  * 4;  // A: 32 rows/pass, 8 f32x4 cols
  const int b_r = t >> 5,  b_c = (t & 31) * 4;  // B: 8 rows/pass, 32 f32x4 cols

  f32x4 ra[4], rb[4];
  auto load_tiles = [&](int k0) {
#pragma unroll
    for (int it = 0; it < 4; ++it)
      ra[it] = *(const f32x4*)&A[(size_t)(m0 + a_r + it * 32) * K + k0 + a_c];
#pragma unroll
    for (int it = 0; it < 4; ++it)
      rb[it] = *(const f32x4*)&Bm[(size_t)(k0 + b_r + it * 8) * N + n0 + b_c];
    if (k0 + 32 < K) {  // warm L2 two tiles ahead (global_prefetch_b8)
      __builtin_prefetch(&A [(size_t)(m0 + a_r) * K + (k0 + 32) + a_c], 0, 3);
      __builtin_prefetch(&Bm[(size_t)(k0 + 32 + b_r) * N + n0 + b_c], 0, 3);
    }
  };

  load_tiles(0);

  for (int k0 = 0; k0 < K; k0 += 32) {
    // Stage current tile regs -> LDS (fp32 -> bf16).
#pragma unroll
    for (int it = 0; it < 4; ++it) {
      int r = a_r + it * 32;
      unsigned int p0 = (unsigned int)f2bf(ra[it][0]) | ((unsigned int)f2bf(ra[it][1]) << 16);
      unsigned int p1 = (unsigned int)f2bf(ra[it][2]) | ((unsigned int)f2bf(ra[it][3]) << 16);
      *(unsigned int*)&As[r][a_c]     = p0;
      *(unsigned int*)&As[r][a_c + 2] = p1;
    }
#pragma unroll
    for (int it = 0; it < 4; ++it) {
      int kr = b_r + it * 8;
#pragma unroll
      for (int i = 0; i < 4; ++i) Bs[b_c + i][kr] = f2bf(rb[it][i]);
    }
    __syncthreads();

    // Kick off next tile's global loads; they overlap the WMMAs below.
    if (k0 + 32 < K) load_tiles(k0 + 32);

    // A frag (16x32 bf16): lane m = l16, elems 0..7 -> k=half*8+e, 8..15 -> +16.
    FragAB a[2], b[4];
#pragma unroll
    for (int mi = 0; mi < 2; ++mi) {
      const unsigned short* rp = &As[wm * 32 + mi * 16 + l16][half * 8];
      a[mi].q[0] = *(const u32x4*)rp;
      a[mi].q[1] = *(const u32x4*)(rp + 16);
    }
    // B frag (32x16 bf16): lane n = l16, elems 0..7 -> k=half*8+e, 8..15 -> +16.
#pragma unroll
    for (int j = 0; j < 4; ++j) {
      const unsigned short* rp = &Bs[wn * 64 + j * 16 + l16][half * 8];
      b[j].q[0] = *(const u32x4*)rp;
      b[j].q[1] = *(const u32x4*)(rp + 16);
    }
#pragma unroll
    for (int mi = 0; mi < 2; ++mi)
#pragma unroll
      for (int j = 0; j < 4; ++j)
        acc[mi][j] = __builtin_amdgcn_wmma_f32_16x16x32_bf16(
            false, a[mi].v, false, b[j].v, (short)0, acc[mi][j], false, false);

    __syncthreads();   // LDS consumed; safe to overwrite next iteration
  }

  // Epilogue. C layout: elem r -> (m = half*8 + r, n = l16).
#pragma unroll
  for (int mi = 0; mi < 2; ++mi) {
#pragma unroll
    for (int j = 0; j < 4; ++j) {
      int gn = n0 + wn * 64 + j * 16 + l16;
      float bv = bias[gn];
#pragma unroll
      for (int r = 0; r < 8; ++r) {
        int gm = m0 + wm * 32 + mi * 16 + half * 8 + r;
        float val = acc[mi][j][r] + bv;
        if (MODE == 0) {
          int which = gn >> 11, rem = gn & (DM - 1);
          int bb = gm >> 11, s = gm & (SEQ - 1);           // gm = bb*SEQ + s
          size_t idx = ((size_t)(bb * NH + (rem >> 7)) * SEQ + s) * DH + (rem & (DH - 1));
          unsigned short* dst = (which == 0) ? qB : (which == 1) ? kB : vB;
          dst[idx] = f2bf(val);
        } else {
          out[(size_t)gm * N + gn] = val;
        }
      }
    }
  }
}

// ---------------------------------------------------------------------------
// RoPE on bf16 Q/K in place. Folds log2(e)/sqrt(DH) into Q so the flash
// kernel can use raw exp2f (hardware v_exp_f32).
// ---------------------------------------------------------------------------
__global__ __launch_bounds__(256) void rope_scale_kernel(
    unsigned short* __restrict__ qB, unsigned short* __restrict__ kB)
{
  int idx = blockIdx.x * 256 + threadIdx.x;     // over B*H*S*(DH/2)
  int d  = idx & 63;
  int s  = (idx >> 6) & (SEQ - 1);
  int bh = idx >> 17;
  size_t base = ((size_t)bh * SEQ + s) * DH;

  float inv_freq = exp2f(-0.20762050593046813f * (float)d); // 10000^(-d/64)
  float fr = (float)s * inv_freq;
  float c = __cosf(fr), sn = __sinf(fr);
  const float QS = 1.4426950408889634f * 0.08838834764831845f; // log2e / sqrt(128)

  float q1 = bf2f(qB[base + d]), q2 = bf2f(qB[base + d + 64]);
  qB[base + d]      = f2bf((q1 * c - q2 * sn) * QS);
  qB[base + d + 64] = f2bf((q1 * sn + q2 * c) * QS);

  float k1 = bf2f(kB[base + d]), k2 = bf2f(kB[base + d + 64]);
  kB[base + d]      = f2bf(k1 * c - k2 * sn);
  kB[base + d + 64] = f2bf(k1 * sn + k2 * c);
}

// ---------------------------------------------------------------------------
// Causal flash attention. Block = 128 threads (4 waves) handles 64 Q rows of
// one (b,h); iterates 64-wide K/V tiles up to the diagonal with online
// softmax. Each wave owns 16 Q rows: scores 4 accs, output 8 accs.
// Q and K tiles are pure byte copies -> global_load_async_to_lds_b128.
// V needs a transpose, so it stays register-staged.
// LDS: Q 17.0KB + K 17.0KB + V(t) 18.0KB + P 9.0KB = 62.4KB.
// ---------------------------------------------------------------------------
__global__ __launch_bounds__(128) void attn_fwd_kernel(
    const unsigned short* __restrict__ qB, const unsigned short* __restrict__ kB,
    const unsigned short* __restrict__ vB, float* __restrict__ attn)
{
  __shared__ __align__(16) unsigned short Qs[64][136];   // [m][d]
  __shared__ __align__(16) unsigned short Ks[64][136];   // [n][d]
  __shared__ __align__(16) unsigned short Vt[128][72];   // transposed: [d][n]
  __shared__ __align__(16) unsigned short Ps[4][16][72]; // per-wave P relayout

  const int t    = threadIdx.x;
  const int lane = t & 31;
  const int w    = t >> 5;
  const int half = lane >> 4;
  const int l16  = lane & 15;
  const int bh   = blockIdx.y;
  const int qi   = blockIdx.x;
  const int q0   = qi * 64;
  const size_t hb = (size_t)bh * SEQ;

  // Q tile (64x128 bf16): async copy straight into LDS.
#pragma unroll
  for (int i = 0; i < 8; ++i) {
    int c = t + i * 128;                 // 0..1023
    int row = c >> 4, c8 = (c & 15) * 8;
    async_load_b128(&Qs[row][c8], &qB[(hb + q0 + row) * DH + c8]);
  }

  const f32x8 zero8 = {0.f,0.f,0.f,0.f,0.f,0.f,0.f,0.f};
  float mi_[8], li_[8];
  f32x8 o[8];
#pragma unroll
  for (int r = 0; r < 8; ++r) { mi_[r] = -3.0e38f; li_[r] = 0.f; }
#pragma unroll
  for (int dj = 0; dj < 8; ++dj) o[dj] = zero8;

  for (int kt = 0; kt <= qi; ++kt) {
    const int k0 = kt * 64;
    __syncthreads();   // previous tile fully consumed
    // K tile: async memory->LDS. V tile: load + transpose through VGPRs.
#pragma unroll
    for (int i = 0; i < 8; ++i) {
      int c = t + i * 128;
      int row = c >> 4, c8 = (c & 15) * 8;
      async_load_b128(&Ks[row][c8], &kB[(hb + k0 + row) * DH + c8]);
      union { u32x4 v; unsigned short u[8]; } vv;
      vv.v = *(const u32x4*)&vB[(hb + k0 + row) * DH + c8];
#pragma unroll
      for (int e = 0; e < 8; ++e) Vt[c8 + e][row] = vv.u[e];
    }
    wait_async0();     // our async copies (incl. Q on first pass) landed
    __syncthreads();

    // S = Q * K^T : B-fragment k-dim is d, so Ks rows give contiguous reads.
    f32x8 sc[4];
#pragma unroll
    for (int j = 0; j < 4; ++j) sc[j] = zero8;
#pragma unroll
    for (int kk = 0; kk < 4; ++kk) {
      FragAB qa;
      const unsigned short* qp = &Qs[w * 16 + l16][kk * 32 + half * 8];
      qa.q[0] = *(const u32x4*)qp;
      qa.q[1] = *(const u32x4*)(qp + 16);
#pragma unroll
      for (int j = 0; j < 4; ++j) {
        FragAB kf;
        const unsigned short* kp = &Ks[j * 16 + l16][kk * 32 + half * 8];
        kf.q[0] = *(const u32x4*)kp;
        kf.q[1] = *(const u32x4*)(kp + 16);
        sc[j] = __builtin_amdgcn_wmma_f32_16x16x32_bf16(
            false, qa.v, false, kf.v, (short)0, sc[j], false, false);
      }
    }

    if (kt == qi) {  // causal mask on the diagonal tile (q0 == k0)
#pragma unroll
      for (int j = 0; j < 4; ++j) {
        int n_g = j * 16 + l16;
#pragma unroll
        for (int r = 0; r < 8; ++r) {
          int m_g = w * 16 + half * 8 + r;
          if (n_g > m_g) sc[j][r] = -3.0e38f;
        }
      }
    }

    // Online softmax. Row m lives across the 16 lanes of one half-wave.
    float alpha[8];
#pragma unroll
    for (int r = 0; r < 8; ++r) {
      float mx = fmaxf(fmaxf(sc[0][r], sc[1][r]), fmaxf(sc[2][r], sc[3][r]));
#pragma unroll
      for (int msk = 1; msk < 16; msk <<= 1) mx = fmaxf(mx, __shfl_xor(mx, msk, 32));
      float mn = fmaxf(mi_[r], mx);
      alpha[r] = exp2f(mi_[r] - mn);
      mi_[r] = mn;
    }
    float rs[8];
#pragma unroll
    for (int r = 0; r < 8; ++r) rs[r] = 0.f;
#pragma unroll
    for (int j = 0; j < 4; ++j)
#pragma unroll
      for (int r = 0; r < 8; ++r) {
        float p = exp2f(sc[j][r] - mi_[r]);
        rs[r] += p;
        // C-layout -> LDS so it can be re-read in A-fragment layout.
        Ps[w][half * 8 + r][j * 16 + l16] = f2bf(p);
      }
#pragma unroll
    for (int r = 0; r < 8; ++r) {
      float v = rs[r];
#pragma unroll
      for (int msk = 1; msk < 16; msk <<= 1) v += __shfl_xor(v, msk, 32);
      li_[r] = li_[r] * alpha[r] + v;
#pragma unroll
      for (int dj = 0; dj < 8; ++dj) o[dj][r] *= alpha[r];
    }

    // O += P * V (K-dim = 64 seq cols -> 2 WMMA steps). Per-wave Ps region,
    // DS ops are in-order within a wave so no barrier is needed here.
#pragma unroll
    for (int kk = 0; kk < 2; ++kk) {
      FragAB pa;
      const unsigned short* pp = &Ps[w][l16][kk * 32 + half * 8];
      pa.q[0] = *(const u32x4*)pp;
      pa.q[1] = *(const u32x4*)(pp + 16);
#pragma unroll
      for (int dj = 0; dj < 8; ++dj) {
        FragAB vf;
        const unsigned short* vp = &Vt[dj * 16 + l16][kk * 32 + half * 8];
        vf.q[0] = *(const u32x4*)vp;
        vf.q[1] = *(const u32x4*)(vp + 16);
        o[dj] = __builtin_amdgcn_wmma_f32_16x16x32_bf16(
            false, pa.v, false, vf.v, (short)0, o[dj], false, false);
      }
    }
  }

  // Finalize: normalize and write fp32 [B, S, H*DH].
  const int b = bh >> 4, h = bh & 15;
#pragma unroll
  for (int r = 0; r < 8; ++r) {
    float inv = 1.0f / li_[r];
    int s = q0 + w * 16 + half * 8 + r;
    float* op = attn + ((size_t)(b * SEQ + s) * DM) + h * DH;
#pragma unroll
    for (int dj = 0; dj < 8; ++dj)
      op[dj * 16 + l16] = o[dj][r] * inv;
  }
}

// ---------------------------------------------------------------------------
extern "C" void kernel_launch(void* const* d_in, const int* in_sizes, int n_in,
                              void* d_out, int out_size, void* d_ws, size_t ws_size,
                              hipStream_t stream) {
  const float* x     = (const float*)d_in[0];
  const float* w_qkv = (const float*)d_in[1];
  const float* b_qkv = (const float*)d_in[2];
  const float* w_out = (const float*)d_in[3];
  const float* b_out = (const float*)d_in[4];
  float* out = (float*)d_out;

  const size_t nElem = (size_t)BATCH * NH * SEQ * DH;   // 8,388,608
  unsigned short* qB = (unsigned short*)d_ws;           // bf16 Q  (16 MB)
  unsigned short* kB = qB + nElem;                      // bf16 K  (16 MB)
  unsigned short* vB = kB + nElem;                      // bf16 V  (16 MB)
  float* attn = (float*)(vB + nElem);                   // fp32 attn out (32 MB)

  // 1) QKV projection + bias, scatter to bf16 Q/K/V in [B,H,S,DH].
  gemm_bf16_wmma<0><<<dim3((3 * DM) / 128, (BATCH * SEQ) / 128), 256, 0, stream>>>(
      x, w_qkv, b_qkv, qB, kB, vB, nullptr, 3 * DM, DM);

  // 2) RoPE in place (+ softmax scale folded into Q).
  rope_scale_kernel<<<(BATCH * NH * SEQ * (DH / 2)) / 256, 256, 0, stream>>>(qB, kB);

  // 3) Causal flash attention.
  attn_fwd_kernel<<<dim3(SEQ / 64, BATCH * NH), 128, 0, stream>>>(qB, kB, vB, attn);

  // 4) Output projection + bias.
  gemm_bf16_wmma<1><<<dim3(DM / 128, (BATCH * SEQ) / 128), 256, 0, stream>>>(
      attn, w_out, b_out, nullptr, nullptr, nullptr, out, DM, DM);
}
